// NetConv_63660005261510
// MI455X (gfx1250) — compile-verified, matrix operand order checked by reading, not run on hardware
//
#include <hip/hip_runtime.h>
#include <hip/hip_bf16.h>
#include <cstdint>
#include <cstddef>

// ---------------------------------------------------------------------------
// MI455X (gfx1250) NetConv: edge MLPs as wave32 WMMA GEMMs.
//   - one wave = 16 edges (M=16), N tiled by 16, K tiled by 32
//   - V_WMMA_F32_16X16X32_F16, f32 accumulators, f16 weights/activations
//   - layer-0 A-fragments gathered straight from global (float4 -> f16)
//   - hidden activations via per-wave LDS tile (16 x 72 f16, conflict-free
//     2x ds_load_b128 fragment reads)
//   - weights prepacked to f16, transposed [Npad x Kpad] (B-frag = 2x b128)
//   - scatter: global_atomic_add_f32 (sum), sign-split int atomics (max)
// ---------------------------------------------------------------------------

#define WPB 8            // waves per block (256 threads)
#define TPB 256
#define ACT_STRIDE 72    // f16 elements per activation row (144B: bank-conflict free)

typedef __attribute__((ext_vector_type(16))) _Float16 v16h;
typedef __attribute__((ext_vector_type(8)))  _Float16 v8h;
typedef __attribute__((ext_vector_type(8)))  float    v8f;
typedef __attribute__((ext_vector_type(4)))  float    v4f;

__device__ __forceinline__ v8f wmma_f16(v16h a, v16h b, v8f c) {
  // 8 args: (neg_a, A, neg_b, B, c_mod, C, reuse_a, reuse_b)
  return __builtin_amdgcn_wmma_f32_16x16x32_f16(false, a, false, b, (short)0, c,
                                                false, false);
}

__device__ __forceinline__ v8f bcast8(float x) {
  v8f r;
#pragma unroll
  for (int i = 0; i < 8; ++i) r[i] = x;
  return r;
}

__device__ __forceinline__ void ld8f(float* o, const float* p) {
  v4f a = *(const v4f*)p;
  v4f b = *(const v4f*)(p + 4);
#pragma unroll
  for (int i = 0; i < 4; ++i) { o[i] = a[i]; o[4 + i] = b[i]; }
}

__device__ __forceinline__ v16h pack_frag(const float* lo, const float* hi) {
  v16h r;
#pragma unroll
  for (int i = 0; i < 8; ++i) {
    r[i]     = (_Float16)lo[i];
    r[8 + i] = (_Float16)hi[i];
  }
  return r;
}

// A-fragment from LDS activation tile (16 rows x ACT_STRIDE f16, row-major).
// ISA 16-bit A 16x32 layout: lane m=L%16,half=L/16; v[0..7] = K {half*8..+7, 16+half*8..+7}
__device__ __forceinline__ v16h afrag_lds(const _Float16* act, int m, int half, int kt) {
  const _Float16* base = act + m * ACT_STRIDE + kt * 32 + half * 8;
  v8h lo = *(const v8h*)(base);
  v8h hi = *(const v8h*)(base + 16);
  v16h r;
#pragma unroll
  for (int i = 0; i < 8; ++i) { r[i] = lo[i]; r[8 + i] = hi[i]; }
  return r;
}

// B-fragment from prepacked transposed weights WT [Npad x Kpad] f16 (global, cache-resident)
__device__ __forceinline__ v16h bfrag(const _Float16* __restrict__ wt, int Kpad,
                                      int nt, int kt, int m, int half) {
  const _Float16* p = wt + (size_t)(nt * 16 + m) * Kpad + kt * 32 + half * 8;
  v8h lo = *(const v8h*)(p);
  v8h hi = *(const v8h*)(p + 16);
  v16h r;
#pragma unroll
  for (int i = 0; i < 8; ++i) { r[i] = lo[i]; r[8 + i] = hi[i]; }
  return r;
}

// 8-feature chunk of the 96-padded edge input [src nf(32) | dst nf(32) | ef(8) | pad]
__device__ __forceinline__ void edge_chunk(float* o, int c, const float* __restrict__ nf,
                                           const float* __restrict__ efp, int s, int d) {
  if (c < 4)       ld8f(o, nf + (size_t)s * 32 + c * 8);
  else if (c < 8)  ld8f(o, nf + (size_t)d * 32 + (c - 4) * 8);
  else if (c == 8) ld8f(o, efp);
  else {
#pragma unroll
    for (int i = 0; i < 8; ++i) o[i] = 0.f;
  }
}

// 8-feature chunk of the reduce input [nf(32) | nfo1(32) | nfo2(32)], nfo2 zero-masked
__device__ __forceinline__ void red_chunk(float* o, int c, const float* __restrict__ nf,
                                          const float* __restrict__ nfo1,
                                          const float* __restrict__ nfo2, int v, float cv) {
  if (c < 4)      ld8f(o, nf + (size_t)v * 32 + c * 8);
  else if (c < 8) ld8f(o, nfo1 + (size_t)v * 32 + (c - 4) * 8);
  else {
    ld8f(o, nfo2 + (size_t)v * 32 + (c - 8) * 8);
    if (!(cv > 0.f)) {
#pragma unroll
      for (int i = 0; i < 8; ++i) o[i] = 0.f;
    }
  }
}

// D-tile (f32 accum) -> LDS activation tile as f16, optional LeakyReLU(0.2).
// C/D layout: lane L holds col n = nt*16 + L%16, rows r + 8*(L/16), r = vgpr 0..7
template <int NT>
__device__ __forceinline__ void store_act(_Float16* act, const v8f* acc, int m, int half,
                                          bool lrelu) {
#pragma unroll
  for (int nt = 0; nt < NT; ++nt) {
#pragma unroll
    for (int r = 0; r < 8; ++r) {
      float v = acc[nt][r];
      if (lrelu) v = v > 0.f ? v : 0.2f * v;
      act[(r + 8 * half) * ACT_STRIDE + nt * 16 + m] = (_Float16)v;
    }
  }
}

// One GEMM layer: act(LDS f16, 16 x K) x WT -> acc (bias-initialized)
template <int NT, int KT>
__device__ __forceinline__ void layer_lds(const _Float16* act, const _Float16* __restrict__ wt,
                                          int Kpad, const float* __restrict__ bias, int nbias,
                                          int m, int half, v8f* acc) {
#pragma unroll
  for (int nt = 0; nt < NT; ++nt) {
    int n = nt * 16 + m;
    acc[nt] = bcast8(n < nbias ? bias[n] : 0.f);
  }
#pragma unroll
  for (int kt = 0; kt < KT; ++kt) {
    v16h a = afrag_lds(act, m, half, kt);
#pragma unroll
    for (int nt = 0; nt < NT; ++nt) {
      v16h b = bfrag(wt, Kpad, nt, kt, m, half);
      acc[nt] = wmma_f16(a, b, acc[nt]);
    }
  }
}

// Layer 0 for edge MLPs: A gathered straight from global (K padded 72 -> 96)
template <int NT>
__device__ __forceinline__ void layer0_edge(const float* __restrict__ nf,
                                            const float* __restrict__ efp, int s, int d,
                                            const _Float16* __restrict__ wt,
                                            const float* __restrict__ bias, int m, int half,
                                            v8f* acc) {
#pragma unroll
  for (int nt = 0; nt < NT; ++nt) acc[nt] = bcast8(bias[nt * 16 + m]);
#pragma unroll
  for (int kt = 0; kt < 3; ++kt) {
    float lo[8], hi[8];
    edge_chunk(lo, kt * 4 + half, nf, efp, s, d);
    edge_chunk(hi, kt * 4 + 2 + half, nf, efp, s, d);
    v16h a = pack_frag(lo, hi);
#pragma unroll
    for (int nt = 0; nt < NT; ++nt) {
      v16h b = bfrag(wt, 96, nt, kt, m, half);
      acc[nt] = wmma_f16(a, b, acc[nt]);
    }
  }
}

// float atomic max via sign-split integer atomics (works for mixed-sign floats
// when the destination is initialized to -inf = 0xFF800000)
__device__ __forceinline__ void atomicMaxF(float* a, float v) {
  if (v >= 0.f) atomicMax((int*)a, __float_as_int(v));
  else          atomicMin((unsigned int*)a, (unsigned int)__float_as_int(v));
}

// ---------------------------------------------------------------------------
// Kernels
// ---------------------------------------------------------------------------

__global__ void init_kernel(float* __restrict__ out, float* __restrict__ nfo1,
                            float* __restrict__ nfo2, float* __restrict__ cnt,
                            long nOut, long n32, long n1) {
  long i = (long)blockIdx.x * blockDim.x + threadIdx.x;
  long stride = (long)gridDim.x * blockDim.x;
  for (long j = i; j < nOut; j += stride) out[j] = 0.f;
  for (long j = i; j < n32; j += stride) {
    nfo1[j] = 0.f;
    ((int*)nfo2)[j] = (int)0xFF800000;  // -inf
  }
  for (long j = i; j < n1; j += stride) cnt[j] = 0.f;
}

// fp32 W [K x N] -> f16 W^T [Npad x Kpad], zero padded
__global__ void prepack_kernel(const float* __restrict__ W, int K, int N,
                               _Float16* __restrict__ WT, int Kpad, int Npad) {
  int idx = blockIdx.x * blockDim.x + threadIdx.x;
  int tot = Npad * Kpad;
  if (idx >= tot) return;
  int n = idx / Kpad;
  int k = idx - n * Kpad;
  float v = (n < N && k < K) ? W[(size_t)k * N + n] : 0.f;
  WT[idx] = (_Float16)v;
}

__global__ void __launch_bounds__(TPB) edge_o2i_kernel(
    const float* __restrict__ nf, const float* __restrict__ ef,
    const int* __restrict__ src, const int* __restrict__ dst,
    float* __restrict__ out, int E,
    const _Float16* __restrict__ w0, const _Float16* __restrict__ w1,
    const _Float16* __restrict__ w2, const _Float16* __restrict__ w3,
    const _Float16* __restrict__ w4,
    const float* __restrict__ b0, const float* __restrict__ b1,
    const float* __restrict__ b2, const float* __restrict__ b3,
    const float* __restrict__ b4) {
  __shared__ __align__(16) _Float16 s_act[WPB][16 * ACT_STRIDE];
  __shared__ int s_dst[WPB][16];
  const int lane = threadIdx.x & 31;
  const int wave = threadIdx.x >> 5;
  const int m = lane & 15, half = lane >> 4;
  long e0 = ((long)blockIdx.x * WPB + wave) * 16;
  long em = e0 + m;
  if (em >= E) em = E - 1;
  const int s = src[em], d = dst[em];
  if (half == 0) s_dst[wave][m] = d;
  const float* efp = ef + em * 8;
  _Float16* act = &s_act[wave][0];

  v8f acc[4];
  layer0_edge<4>(nf, efp, s, d, w0, b0, m, half, acc);
  store_act<4>(act, acc, m, half, true);
  layer_lds<4, 2>(act, w1, 64, b1, 64, m, half, acc);
  store_act<4>(act, acc, m, half, true);
  layer_lds<4, 2>(act, w2, 64, b2, 64, m, half, acc);
  store_act<4>(act, acc, m, half, true);
  layer_lds<4, 2>(act, w3, 64, b3, 64, m, half, acc);
  store_act<4>(act, acc, m, half, true);
  layer_lds<4, 2>(act, w4, 64, b4, 64, m, half, acc);  // final: no activation

  // sum-scatter rows to out[dst]
#pragma unroll
  for (int nt = 0; nt < 4; ++nt) {
    const int col = nt * 16 + m;
#pragma unroll
    for (int r = 0; r < 8; ++r) {
      const int row = r + 8 * half;
      if (e0 + row < E) {
        const int dn = s_dst[wave][row];
        atomicAdd(out + (size_t)dn * 64 + col, acc[nt][r]);
      }
    }
  }
}

__global__ void __launch_bounds__(TPB) edge_i2o_kernel(
    const float* __restrict__ nf, const float* __restrict__ ef,
    const int* __restrict__ src, const int* __restrict__ dst,
    float* __restrict__ nfo1, float* __restrict__ nfo2, float* __restrict__ cnt, int E,
    const _Float16* __restrict__ w0, const _Float16* __restrict__ w1,
    const _Float16* __restrict__ w2, const _Float16* __restrict__ w3,
    const float* __restrict__ b0, const float* __restrict__ b1,
    const float* __restrict__ b2, const float* __restrict__ b3) {
  __shared__ __align__(16) _Float16 s_act[WPB][16 * ACT_STRIDE];
  __shared__ int s_dst[WPB][16];
  __shared__ float s_gate[WPB][16];
  const int lane = threadIdx.x & 31;
  const int wave = threadIdx.x >> 5;
  const int m = lane & 15, half = lane >> 4;
  long e0 = ((long)blockIdx.x * WPB + wave) * 16;
  long em = e0 + m;
  if (em >= E) em = E - 1;
  const int s = src[em], d = dst[em];
  if (half == 0) s_dst[wave][m] = d;
  const float* efp = ef + em * 8;
  _Float16* act = &s_act[wave][0];

  v8f acc[4];
  layer0_edge<4>(nf, efp, s, d, w0, b0, m, half, acc);
  store_act<4>(act, acc, m, half, true);
  layer_lds<4, 2>(act, w1, 64, b1, 64, m, half, acc);
  store_act<4>(act, acc, m, half, true);
  layer_lds<4, 2>(act, w2, 64, b2, 64, m, half, acc);
  store_act<4>(act, acc, m, half, true);

  // final layer: 64 -> 65 real outputs (N padded to 80 -> 5 tiles)
  v8f acc5[5];
  layer_lds<5, 2>(act, w3, 64, b3, 65, m, half, acc5);

  // gate column (col 0) -> sigmoid -> broadcast per row via LDS
  if (m == 0) {
#pragma unroll
    for (int r = 0; r < 8; ++r) {
      float g = acc5[0][r];
      s_gate[wave][r + 8 * half] = 1.f / (1.f + __expf(-g));
    }
  }
  asm volatile("s_wait_dscnt 0x0" ::: "memory");
  float kk[8];
#pragma unroll
  for (int r = 0; r < 8; ++r) kk[r] = s_gate[wave][r + 8 * half];

#pragma unroll
  for (int nt = 0; nt < 5; ++nt) {
    const int col = nt * 16 + m;
#pragma unroll
    for (int r = 0; r < 8; ++r) {
      const int row = r + 8 * half;
      if (e0 + row >= E) continue;
      const int dn = s_dst[wave][row];
      const float v = acc5[nt][r] * kk[r];
      if (col == 0) {
        atomicAdd(cnt + dn, 1.f);                                  // in-degree count
      } else if (col <= 32) {
        atomicAdd(nfo1 + (size_t)dn * 32 + (col - 1), v);          // f1 sum
      } else if (col <= 64) {
        atomicMaxF(nfo2 + (size_t)dn * 32 + (col - 33), v);        // f2 max
      }                                                            // col >= 65: pad
    }
  }
}

__global__ void __launch_bounds__(TPB) reduce_kernel(
    const float* __restrict__ nf, const float* __restrict__ nfo1,
    const float* __restrict__ nfo2, const float* __restrict__ cnt,
    const int* __restrict__ onodes, float* __restrict__ out, int NOUT,
    const _Float16* __restrict__ w0, const _Float16* __restrict__ w1,
    const _Float16* __restrict__ w2, const _Float16* __restrict__ w3,
    const float* __restrict__ b0, const float* __restrict__ b1,
    const float* __restrict__ b2, const float* __restrict__ b3) {
  __shared__ __align__(16) _Float16 s_act[WPB][16 * ACT_STRIDE];
  __shared__ int s_v[WPB][16];
  const int lane = threadIdx.x & 31;
  const int wave = threadIdx.x >> 5;
  const int m = lane & 15, half = lane >> 4;
  long i0 = ((long)blockIdx.x * WPB + wave) * 16;
  long im = i0 + m;
  if (im >= NOUT) im = NOUT - 1;
  const int v = onodes[im];
  if (half == 0) s_v[wave][m] = v;
  const float cv = cnt[v];
  _Float16* act = &s_act[wave][0];

  v8f acc[4];
  // layer 0: gathered [nf | nfo1 | masked nfo2], K = 96 exactly
#pragma unroll
  for (int nt = 0; nt < 4; ++nt) acc[nt] = bcast8(b0[nt * 16 + m]);
#pragma unroll
  for (int kt = 0; kt < 3; ++kt) {
    float lo[8], hi[8];
    red_chunk(lo, kt * 4 + half, nf, nfo1, nfo2, v, cv);
    red_chunk(hi, kt * 4 + 2 + half, nf, nfo1, nfo2, v, cv);
    v16h a = pack_frag(lo, hi);
#pragma unroll
    for (int nt = 0; nt < 4; ++nt) {
      v16h b = bfrag(w0, 96, nt, kt, m, half);
      acc[nt] = wmma_f16(a, b, acc[nt]);
    }
  }
  store_act<4>(act, acc, m, half, true);
  layer_lds<4, 2>(act, w1, 64, b1, 64, m, half, acc);
  store_act<4>(act, acc, m, half, true);
  layer_lds<4, 2>(act, w2, 64, b2, 64, m, half, acc);
  store_act<4>(act, acc, m, half, true);
  layer_lds<4, 2>(act, w3, 64, b3, 64, m, half, acc);  // final: no activation

  // overwrite output rows (runs after o2i scatter in stream order)
#pragma unroll
  for (int nt = 0; nt < 4; ++nt) {
    const int col = nt * 16 + m;
#pragma unroll
    for (int r = 0; r < 8; ++r) {
      const int row = r + 8 * half;
      if (i0 + row < NOUT) {
        const int vv = s_v[wave][row];
        out[(size_t)vv * 64 + col] = acc[nt][r];
      }
    }
  }
}

// ---------------------------------------------------------------------------
// Host
// ---------------------------------------------------------------------------

extern "C" void kernel_launch(void* const* d_in, const int* in_sizes, int n_in,
                              void* d_out, int out_size, void* d_ws, size_t ws_size,
                              hipStream_t stream) {
  const float* nf      = (const float*)d_in[0];
  const float* ef_out  = (const float*)d_in[1];
  const float* ef_in   = (const float*)d_in[2];
  const int*   src_out = (const int*)d_in[3];
  const int*   dst_out = (const int*)d_in[4];
  const int*   src_in  = (const int*)d_in[5];
  const int*   dst_in  = (const int*)d_in[6];
  const int*   onodes  = (const int*)d_in[7];
  const int N    = in_sizes[0] / 32;
  const int E    = in_sizes[3];
  const int NOUT = in_sizes[7];
  (void)n_in; (void)out_size; (void)ws_size;

  // Detect param group order: insertion order (msg_o2i first) vs
  // sorted-dict order (msg_i2o first). msg_i2o layer-3 W has 64*65 = 4160.
  int pA = 8, pB = 18, pC = 26;  // A = msg_o2i(5 layers), B = msg_i2o(4), C = reduce_o(4)
  if (in_sizes[14] == 4160) { pB = 8; pA = 16; pC = 26; }

  const float *WA[5], *BA[5], *WB[4], *BB[4], *WC[4], *BC[4];
  for (int l = 0; l < 5; ++l) { WA[l] = (const float*)d_in[pA + 2 * l]; BA[l] = (const float*)d_in[pA + 2 * l + 1]; }
  for (int l = 0; l < 4; ++l) { WB[l] = (const float*)d_in[pB + 2 * l]; BB[l] = (const float*)d_in[pB + 2 * l + 1]; }
  for (int l = 0; l < 4; ++l) { WC[l] = (const float*)d_in[pC + 2 * l]; BC[l] = (const float*)d_in[pC + 2 * l + 1]; }

  // workspace carve-out
  uint8_t* ws = (uint8_t*)d_ws;
  size_t off = 0;
  auto take = [&](size_t bytes) -> uint8_t* {
    uint8_t* p = ws + off;
    off += (bytes + 255) & ~(size_t)255;
    return p;
  };
  float* nfo1 = (float*)take((size_t)N * 32 * sizeof(float));
  float* nfo2 = (float*)take((size_t)N * 32 * sizeof(float));
  float* cntb = (float*)take((size_t)N * sizeof(float));
  _Float16* wtA[5];
  wtA[0] = (_Float16*)take(64 * 96 * 2);
  for (int l = 1; l < 5; ++l) wtA[l] = (_Float16*)take(64 * 64 * 2);
  _Float16* wtB[4];
  wtB[0] = (_Float16*)take(64 * 96 * 2);
  wtB[1] = (_Float16*)take(64 * 64 * 2);
  wtB[2] = (_Float16*)take(64 * 64 * 2);
  wtB[3] = (_Float16*)take(80 * 64 * 2);
  _Float16* wtC[4];
  wtC[0] = (_Float16*)take(64 * 96 * 2);
  for (int l = 1; l < 4; ++l) wtC[l] = (_Float16*)take(64 * 64 * 2);

  auto pp = [&](const float* W, int K, int Nn, _Float16* WT, int Kpad, int Npad) {
    int tot = Npad * Kpad;
    prepack_kernel<<<(tot + 255) / 256, 256, 0, stream>>>(W, K, Nn, WT, Kpad, Npad);
  };
  pp(WA[0], 72, 64, wtA[0], 96, 64);
  for (int l = 1; l < 5; ++l) pp(WA[l], 64, 64, wtA[l], 64, 64);
  pp(WB[0], 72, 64, wtB[0], 96, 64);
  pp(WB[1], 64, 64, wtB[1], 64, 64);
  pp(WB[2], 64, 64, wtB[2], 64, 64);
  pp(WB[3], 64, 65, wtB[3], 64, 80);
  pp(WC[0], 96, 64, wtC[0], 96, 64);
  for (int l = 1; l < 4; ++l) pp(WC[l], 64, 64, wtC[l], 64, 64);

  init_kernel<<<4096, 256, 0, stream>>>((float*)d_out, nfo1, nfo2, cntb,
                                        (long)N * 64, (long)N * 32, (long)N);

  const int edge_blocks = (E + (WPB * 16) - 1) / (WPB * 16);
  edge_o2i_kernel<<<edge_blocks, TPB, 0, stream>>>(
      nf, ef_out, src_out, dst_out, (float*)d_out, E,
      wtA[0], wtA[1], wtA[2], wtA[3], wtA[4], BA[0], BA[1], BA[2], BA[3], BA[4]);

  edge_i2o_kernel<<<edge_blocks, TPB, 0, stream>>>(
      nf, ef_in, src_in, dst_in, nfo1, nfo2, cntb, E,
      wtB[0], wtB[1], wtB[2], wtB[3], BB[0], BB[1], BB[2], BB[3]);

  const int red_blocks = (NOUT + (WPB * 16) - 1) / (WPB * 16);
  reduce_kernel<<<red_blocks, TPB, 0, stream>>>(
      nf, nfo1, nfo2, cntb, onodes, (float*)d_out, NOUT,
      wtC[0], wtC[1], wtC[2], wtC[3], BC[0], BC[1], BC[2], BC[3]);
}
